// TransitionUp_15917148799055
// MI455X (gfx1250) — matrix operand order repaired and checked by reading, not compiled, not run on hardware
//
#include <hip/hip_runtime.h>
#include <hip/hip_bf16.h>

// ---------------- Problem constants (from reference) ----------------
#define CN   256      // channels C
#define BD   16       // batch segments B
#define D3N  1024     // f3 width
#define EPSV 1e-5f
#define BPACK_ELEMS (8 * 16 * 32 * 16)   // 65536 bf16 = 128 KB

typedef __attribute__((ext_vector_type(8)))  float  v8f;
typedef __attribute__((ext_vector_type(16))) __bf16 v16bf;
typedef __attribute__((ext_vector_type(4)))  int    v4i;

// Workspace layout (floats), zero-region first
#define F_SEGSUM 0                 // B*C   = 4096
#define F_S1     4096              // C     = 256
#define F_S2     4352              // C     = 256
#define ZERO_FLOATS 4608
#define F_BIAS   4608              // B*C   = 4096
#define F_SCALE  8704              // C     = 256
#define F_SHIFT  8960              // C     = 256
#define BPACK_BYTE_OFF 36864       // 9216 floats * 4; 32B aligned; then 128 KB bf16

// ---------------- K0: per-segment sums of x ----------------
__global__ void seg_x_sum(const float* __restrict__ x, const int* __restrict__ o,
                          float* __restrict__ segsum, int N) {
  __shared__ int oS[BD];
  int c = threadIdx.x;                   // one channel per thread
  if (c < BD) oS[c] = o[c];
  __syncthreads();
  int r0 = blockIdx.x * 64;
  int cur = 0;
  while (cur < BD - 1 && r0 >= oS[cur]) cur++;
  float acc = 0.f;
  for (int i = 0; i < 64; ++i) {
    int r = r0 + i;
    if (r >= N) break;
    int s = cur;
    while (s < BD - 1 && r >= oS[s]) s++;
    if (s != cur) { atomicAdd(&segsum[cur * CN + c], acc); acc = 0.f; cur = s; }
    acc += x[(size_t)r * CN + c];
  }
  atomicAdd(&segsum[cur * CN + c], acc);
}

// ---------------- K1: means -> f2, f3 -> per-segment bias table ----------------
__global__ void small_mlp(const float* __restrict__ segsum, const int* __restrict__ o,
                          const int* __restrict__ y,
                          const float* __restrict__ W1, const float* __restrict__ b1,
                          const float* __restrict__ W2, const float* __restrict__ b2,
                          const float* __restrict__ W3, const float* __restrict__ b3,
                          float* __restrict__ biasOut) {
  __shared__ float meanS[CN];
  __shared__ float f2S[CN];
  __shared__ float f3S[D3N];
  int b = blockIdx.x, t = threadIdx.x;
  int lo = (b == 0) ? 0 : o[b - 1];
  float cnt = (float)(o[b] - lo);
  meanS[t] = segsum[b * CN + t] / cnt;
  __syncthreads();
  // f2 = relu(mean @ W2 + b2)
  float a2 = b2[t];
  for (int k = 0; k < CN; ++k) a2 += meanS[k] * W2[k * CN + t];
  f2S[t] = fmaxf(a2, 0.f);
  // f3 = relu(W3[y] + b3)
  int cls = y[b];
  for (int d = t; d < D3N; d += CN) f3S[d] = fmaxf(W3[cls * D3N + d] + b3[d], 0.f);
  __syncthreads();
  // bias = f2 @ W1[256:512] + f3 @ W1[512:1536] + b1
  float bb = b1[t];
  for (int j = 0; j < CN; ++j)  bb += f2S[j] * W1[(CN + j) * CN + t];
  for (int j = 0; j < D3N; ++j) bb += f3S[j] * W1[(2 * CN + j) * CN + t];
  biasOut[b * CN + t] = bb;
}

// ---------------- Kp: pack W1[0:256][:] into bf16 WMMA B-fragments ----------------
__global__ void pack_w1(const float* __restrict__ W1, __bf16* __restrict__ Bp) {
  int t = blockIdx.x * 256 + threadIdx.x;   // 65536 total
  int e = t & 15;
  int L = (t >> 4) & 31;
  int frag = t >> 9;
  int nb = frag & 15;
  int kb = frag >> 4;
  int k = kb * 32 + (L >> 4) * 8 + (e & 7) + (e >> 3) * 16;
  int n = nb * 16 + (L & 15);
  Bp[t] = (__bf16)W1[k * CN + n];
}

// ---------------- Shared GEMM tile: 16 rows x 256 cols per wave ----------------
// B fragments are read from LDS (ds_load) so the per-WMMA dependency wait is a
// short dscnt wait, not an L2 round-trip.
__device__ __forceinline__ void gemm_tile(const float* __restrict__ x,
                                          const __bf16* BpS,   // LDS
                                          int rowbase, int lane, v8f acc[16]) {
  const int m = rowbase + (lane & 15);
  const int half = lane >> 4;
  const float* xbase = x + (size_t)m * CN + half * 8;
  #pragma unroll
  for (int nb = 0; nb < 16; ++nb) {
    v8f z = {0.f, 0.f, 0.f, 0.f, 0.f, 0.f, 0.f, 0.f};
    acc[nb] = z;
  }
  // A operand prefetched one K-step (16 WMMAs) ahead to cover HBM latency.
  v8f lo = *(const v8f*)(xbase);
  v8f hi = *(const v8f*)(xbase + 16);
  #pragma unroll
  for (int kb = 0; kb < 8; ++kb) {
    v16bf a;
    #pragma unroll
    for (int i = 0; i < 8; ++i) { a[i] = (__bf16)lo[i]; a[8 + i] = (__bf16)hi[i]; }
    if (kb < 7) {
      lo = *(const v8f*)(xbase + (kb + 1) * 32);
      hi = *(const v8f*)(xbase + (kb + 1) * 32 + 16);
    }
    #pragma unroll
    for (int nb = 0; nb < 16; ++nb) {
      v16bf bf = *(const v16bf*)(BpS + (((kb * 16 + nb) * 32 + lane) << 4));
      acc[nb] = __builtin_amdgcn_wmma_f32_16x16x32_bf16(
          false, a, false, bf, (short)0, acc[nb], false, false);
    }
  }
}

// cooperative bulk copy of the 128 KB weight pack into LDS
__device__ __forceinline__ void stage_bpack(const __bf16* __restrict__ Bp,
                                            __bf16* BpS, int tid) {
  const v4i* src = (const v4i*)Bp;       // 16B chunks
  v4i* dst = (v4i*)BpS;
  #pragma unroll
  for (int i = 0; i < (BPACK_ELEMS / 8) / 256; ++i)   // 32 iters
    dst[tid + i * 256] = src[tid + i * 256];
}

// ---------------- K2: GEMM pass 1 -> per-channel sum / sumsq of h ----------------
__global__ void __launch_bounds__(256, 1)
gemm_stats(const float* __restrict__ x, const __bf16* __restrict__ Bp,
           const float* __restrict__ bias, const int* __restrict__ o,
           float* __restrict__ gs1, float* __restrict__ gs2) {
  __shared__ __align__(16) __bf16 BpS[BPACK_ELEMS];   // 128 KB
  __shared__ float biasS[BD * CN];
  __shared__ float s1S[CN], s2S[CN];
  __shared__ int oS[BD];
  int tid = threadIdx.x;
  stage_bpack(Bp, BpS, tid);
  for (int i = tid; i < BD * CN; i += 256) biasS[i] = bias[i];
  s1S[tid] = 0.f; s2S[tid] = 0.f;
  if (tid < BD) oS[tid] = o[tid];
  __syncthreads();

  int wave = tid >> 5, lane = tid & 31;
  int rowbase = blockIdx.x * 128 + wave * 16;
  v8f acc[16];
  gemm_tile(x, BpS, rowbase, lane, acc);

  int half = lane >> 4;
  int r0 = rowbase + half * 8;
  int segs[8];
  {
    int s = 0;
    #pragma unroll
    for (int e = 0; e < 8; ++e) {
      int r = r0 + e;
      while (s < BD - 1 && r >= oS[s]) s++;
      segs[e] = s;
    }
  }
  int cn = lane & 15;
  #pragma unroll
  for (int nb = 0; nb < 16; ++nb) {
    int c = nb * 16 + cn;
    float s1 = 0.f, s2 = 0.f;
    #pragma unroll
    for (int e = 0; e < 8; ++e) {
      float h = acc[nb][e] + biasS[segs[e] * CN + c];
      s1 += h; s2 += h * h;
    }
    atomicAdd(&s1S[c], s1);
    atomicAdd(&s2S[c], s2);
  }
  __syncthreads();
  atomicAdd(&gs1[tid], s1S[tid]);
  atomicAdd(&gs2[tid], s2S[tid]);
}

// ---------------- K3: fold BN stats + gamma/beta into scale/shift ----------------
__global__ void finalize_stats(const float* __restrict__ gs1, const float* __restrict__ gs2,
                               const float* __restrict__ gamma, const float* __restrict__ beta,
                               float* __restrict__ scale, float* __restrict__ shift, float invN) {
  int c = threadIdx.x;
  float mu  = gs1[c] * invN;
  float var = gs2[c] * invN - mu * mu;
  float inv = rsqrtf(var + EPSV);
  float sc  = gamma[c] * inv;
  scale[c] = sc;
  shift[c] = beta[c] - sc * mu;
}

// ---------------- K4: GEMM pass 2 -> normalize + ReLU -> out ----------------
__global__ void __launch_bounds__(256, 1)
gemm_out(const float* __restrict__ x, const __bf16* __restrict__ Bp,
         const float* __restrict__ bias, const int* __restrict__ o,
         const float* __restrict__ scale, const float* __restrict__ shift,
         float* __restrict__ out) {
  __shared__ __align__(16) __bf16 BpS[BPACK_ELEMS];   // 128 KB
  __shared__ float biasS[BD * CN];
  __shared__ float scS[CN], shS[CN];
  __shared__ int oS[BD];
  int tid = threadIdx.x;
  stage_bpack(Bp, BpS, tid);
  for (int i = tid; i < BD * CN; i += 256) biasS[i] = bias[i];
  scS[tid] = scale[tid]; shS[tid] = shift[tid];
  if (tid < BD) oS[tid] = o[tid];
  __syncthreads();

  int wave = tid >> 5, lane = tid & 31;
  int rowbase = blockIdx.x * 128 + wave * 16;
  v8f acc[16];
  gemm_tile(x, BpS, rowbase, lane, acc);

  int half = lane >> 4;
  int r0 = rowbase + half * 8;
  int segs[8];
  {
    int s = 0;
    #pragma unroll
    for (int e = 0; e < 8; ++e) {
      int r = r0 + e;
      while (s < BD - 1 && r >= oS[s]) s++;
      segs[e] = s;
    }
  }
  int cn = lane & 15;
  #pragma unroll
  for (int nb = 0; nb < 16; ++nb) {
    int c = nb * 16 + cn;
    float sc = scS[c], sh = shS[c];
    #pragma unroll
    for (int e = 0; e < 8; ++e) {
      float h = acc[nb][e] + biasS[segs[e] * CN + c];
      out[(size_t)(r0 + e) * CN + c] = fmaxf(fmaf(h, sc, sh), 0.f);
    }
  }
}

// ---------------- host launcher ----------------
extern "C" void kernel_launch(void* const* d_in, const int* in_sizes, int n_in,
                              void* d_out, int out_size, void* d_ws, size_t ws_size,
                              hipStream_t stream) {
  const float* x     = (const float*)d_in[0];
  const int*   o     = (const int*)d_in[1];
  const int*   y     = (const int*)d_in[2];
  const float* W1    = (const float*)d_in[3];
  const float* b1    = (const float*)d_in[4];
  const float* gamma = (const float*)d_in[5];
  const float* beta  = (const float*)d_in[6];
  const float* W2    = (const float*)d_in[7];
  const float* b2    = (const float*)d_in[8];
  const float* W3    = (const float*)d_in[9];
  const float* b3    = (const float*)d_in[10];

  const int N = in_sizes[0] / CN;        // 131072
  float*  ws = (float*)d_ws;
  __bf16* Bp = (__bf16*)((char*)d_ws + BPACK_BYTE_OFF);

  // zero the accumulators (segsum, s1, s2) every call — deterministic
  hipMemsetAsync(ws, 0, ZERO_FLOATS * sizeof(float), stream);

  seg_x_sum<<<(N + 63) / 64, 256, 0, stream>>>(x, o, ws + F_SEGSUM, N);
  pack_w1<<<BPACK_ELEMS / 256, 256, 0, stream>>>(W1, Bp);
  small_mlp<<<BD, CN, 0, stream>>>(ws + F_SEGSUM, o, y, W1, b1, W2, b2, W3, b3,
                                   ws + F_BIAS);
  gemm_stats<<<N / 128, 256, 0, stream>>>(x, Bp, ws + F_BIAS, o,
                                          ws + F_S1, ws + F_S2);
  finalize_stats<<<1, CN, 0, stream>>>(ws + F_S1, ws + F_S2, gamma, beta,
                                       ws + F_SCALE, ws + F_SHIFT, 1.0f / (float)N);
  gemm_out<<<N / 128, 256, 0, stream>>>(x, Bp, ws + F_BIAS, o,
                                        ws + F_SCALE, ws + F_SHIFT, (float*)d_out);
}
